// SRU_44229573214889
// MI455X (gfx1250) — matrix operand order repaired and checked by reading, not compiled, not run on hardware
//
#include <hip/hip_runtime.h>
#include <hip/hip_bf16.h>
#include <stdint.h>

// Problem constants (from the reference)
#define B_    64
#define T_    512
#define I_    512
#define H_    1024
#define G3H   3072          // 3*H
#define MROWS (B_ * T_)     // 32768

typedef __attribute__((ext_vector_type(16))) __bf16 v16bf;
typedef __attribute__((ext_vector_type(8)))  float  v8f;

union FragAB { uint4 q[2]; v16bf bf; };
union Acc    { v8f v; float f[8]; };

__device__ __forceinline__ unsigned short f2bf(float f) {
  unsigned u = __float_as_uint(f);
  return (unsigned short)((u + 0x7FFFu + ((u >> 16) & 1u)) >> 16);  // RNE
}
__device__ __forceinline__ float bf2f(unsigned short h) {
  return __uint_as_float(((unsigned)h) << 16);
}
__device__ __forceinline__ float sigmoidf_(float x) {
  return 1.0f / (1.0f + __expf(-x));
}
__device__ __forceinline__ float tanhf_(float x) {
  return 2.0f * sigmoidf_(2.0f * x) - 1.0f;
}

// ---------------------------------------------------------------- utilities
__global__ void k_init_ctr(unsigned* ctr) {
  if (threadIdx.x < 16) ctr[threadIdx.x] = 0;
}

__global__ void k_f32_to_bf16(const float* __restrict__ src,
                              unsigned short* __restrict__ dst, int n) {
  int i = blockIdx.x * blockDim.x + threadIdx.x;
  int stride = gridDim.x * blockDim.x;
  for (; i < n; i += stride) dst[i] = f2bf(src[i]);
}

// ------------------------------------------------- input-side GEMM (WMMA)
// out[M,3072] = A[M,K] @ W[3072,K]^T + bias ; all bf16 in, f32 accumulate.
// WG = 256 threads = 8 waves (4 M x 2 N). Wave tile = 32x64 (2x4 WMMA tiles).
__global__ __launch_bounds__(256) void k_gemm_gx(
    const unsigned short* __restrict__ A,
    const unsigned short* __restrict__ W,
    const float* __restrict__ bias,
    unsigned short* __restrict__ out,
    int K)
{
  const int lane  = threadIdx.x & 31;
  const int nlane = lane & 15;
  const int u     = lane >> 4;
  const int wave  = threadIdx.x >> 5;
  const int wm    = wave & 3;
  const int wn    = wave >> 2;
  const int mbase = blockIdx.x * 128 + wm * 32;
  const int nbase = blockIdx.y * 128 + wn * 64;

  Acc acc[2][4];
  #pragma unroll
  for (int i = 0; i < 2; ++i)
    #pragma unroll
    for (int j = 0; j < 4; ++j)
      #pragma unroll
      for (int v = 0; v < 8; ++v) acc[i][j].f[v] = 0.0f;

  float bs[4];
  #pragma unroll
  for (int nt = 0; nt < 4; ++nt) bs[nt] = bias[nbase + nt * 16 + nlane];

  for (int k = 0; k < K; k += 32) {
    FragAB a[2];
    #pragma unroll
    for (int mt = 0; mt < 2; ++mt) {
      const unsigned short* ap =
          A + (size_t)(mbase + mt * 16 + nlane) * K + k + u * 16;
      a[mt].q[0] = *(const uint4*)ap;
      a[mt].q[1] = *(const uint4*)(ap + 8);
    }
    #pragma unroll
    for (int nt = 0; nt < 4; ++nt) {
      FragAB b;
      const unsigned short* bp =
          W + (size_t)(nbase + nt * 16 + nlane) * K + k + u * 16;
      b.q[0] = *(const uint4*)bp;
      b.q[1] = *(const uint4*)(bp + 8);
      acc[0][nt].v = __builtin_amdgcn_wmma_f32_16x16x32_bf16(
          false, a[0].bf, false, b.bf, (short)0, acc[0][nt].v, false, false);
      acc[1][nt].v = __builtin_amdgcn_wmma_f32_16x16x32_bf16(
          false, a[1].bf, false, b.bf, (short)0, acc[1][nt].v, false, false);
    }
  }

  #pragma unroll
  for (int mt = 0; mt < 2; ++mt)
    #pragma unroll
    for (int nt = 0; nt < 4; ++nt) {
      int col = nbase + nt * 16 + nlane;
      #pragma unroll
      for (int v = 0; v < 8; ++v) {
        int row = mbase + mt * 16 + v + 8 * u;
        out[(size_t)row * G3H + col] = f2bf(acc[mt][nt].f[v] + bs[nt]);
      }
    }
}

// ------------------------------------------ global barrier (per batch group)
// Monotonic counter, 16 WGs per group. All WGs resident (64 tiny WGs total).
__device__ __forceinline__ void group_barrier(unsigned* ctr, unsigned target) {
  __threadfence();          // release own stores to device scope
  __syncthreads();
  if (threadIdx.x == 0) {
    atomicAdd(ctr, 1u);
    volatile unsigned* vc = (volatile unsigned*)ctr;
    while (*vc < target) __builtin_amdgcn_s_sleep(1);
    __threadfence();        // acquire: invalidate stale WGP$ lines
  }
  __syncthreads();
}

// -------------------------------------------------- recurrent scan (WMMA)
// grid = 64 WGs: blockIdx = g*16 + c ; g = batch group (16 rows), c = 64-col
// slice of H. WG = 128 threads = 4 waves; wave w owns h-cols [c*64+16w, +16)
// and its 3 gate tiles (r,z,n). Per step: 3x32 = 96 wmma per wave, then the
// fp32 gate math in registers, bf16 h ping-pong store, one global barrier.
__global__ __launch_bounds__(128) void k_gru_scan(
    const unsigned short* __restrict__ gx,    // [B*T, 3072] bf16 (bih folded)
    const unsigned short* __restrict__ Whh,   // [3072, 1024] bf16
    const float* __restrict__ bhh,            // [3072]
    unsigned short* __restrict__ hping,       // [2][64][1024] bf16
    unsigned short* __restrict__ seqOut,      // [B*T, 1024] bf16
    float* __restrict__ hFinal,               // [64, 1024] f32
    unsigned* __restrict__ ctr)               // 4 counters (one per group)
{
  const int g     = blockIdx.x >> 4;
  const int c     = blockIdx.x & 15;
  const int lane  = threadIdx.x & 31;
  const int nlane = lane & 15;
  const int u     = lane >> 4;
  const int w     = threadIdx.x >> 5;
  const int colh  = c * 64 + w * 16 + nlane;   // 0..1023
  unsigned* myctr = ctr + g;

  const float bhR = bhh[colh];
  const float bhZ = bhh[H_ + colh];
  const float bhN = bhh[2 * H_ + colh];

  const unsigned short* WR = Whh + (size_t)colh * H_;
  const unsigned short* WZ = Whh + (size_t)(H_ + colh) * H_;
  const unsigned short* WN = Whh + (size_t)(2 * H_ + colh) * H_;

  float hReg[8];
  #pragma unroll
  for (int v = 0; v < 8; ++v) hReg[v] = 0.0f;

  // zero-init ping buffer 0 (owner lanes only), then sync the group
  #pragma unroll
  for (int v = 0; v < 8; ++v) {
    int b = g * 16 + v + 8 * u;
    hping[(size_t)b * H_ + colh] = 0;
  }
  unsigned epoch = 1;
  group_barrier(myctr, 16u * epoch);

  for (int t = 0; t < T_; ++t) {
    const unsigned short* hRd = hping + (size_t)(t & 1) * (B_ * H_);
    unsigned short*       hWr = hping + (size_t)((t + 1) & 1) * (B_ * H_);

    Acc aR, aZ, aN;
    #pragma unroll
    for (int v = 0; v < 8; ++v) { aR.f[v] = 0.f; aZ.f[v] = 0.f; aN.f[v] = 0.f; }

    const unsigned short* hRow = hRd + (size_t)(g * 16 + nlane) * H_;
    for (int k = 0; k < H_; k += 32) {
      FragAB fa, fr, fz, fn;
      fa.q[0] = *(const uint4*)(hRow + k + u * 16);
      fa.q[1] = *(const uint4*)(hRow + k + u * 16 + 8);
      fr.q[0] = *(const uint4*)(WR + k + u * 16);
      fr.q[1] = *(const uint4*)(WR + k + u * 16 + 8);
      fz.q[0] = *(const uint4*)(WZ + k + u * 16);
      fz.q[1] = *(const uint4*)(WZ + k + u * 16 + 8);
      fn.q[0] = *(const uint4*)(WN + k + u * 16);
      fn.q[1] = *(const uint4*)(WN + k + u * 16 + 8);
      aR.v = __builtin_amdgcn_wmma_f32_16x16x32_bf16(
          false, fa.bf, false, fr.bf, (short)0, aR.v, false, false);
      aZ.v = __builtin_amdgcn_wmma_f32_16x16x32_bf16(
          false, fa.bf, false, fz.bf, (short)0, aZ.v, false, false);
      aN.v = __builtin_amdgcn_wmma_f32_16x16x32_bf16(
          false, fa.bf, false, fn.bf, (short)0, aN.v, false, false);
    }

    #pragma unroll
    for (int v = 0; v < 8; ++v) {
      int b = g * 16 + v + 8 * u;                 // C/D layout: M = v + 8u
      size_t gxo = ((size_t)b * T_ + t) * G3H;
      float xr = bf2f(gx[gxo + colh]);
      float xz = bf2f(gx[gxo + H_ + colh]);
      float xn = bf2f(gx[gxo + 2 * H_ + colh]);
      if (t + 1 < T_) {                            // warm next step's gx lines
        __builtin_prefetch(gx + gxo + G3H + colh, 0, 1);
        __builtin_prefetch(gx + gxo + G3H + 2 * H_ + colh, 0, 1);
      }
      float r  = sigmoidf_(xr + aR.f[v] + bhR);
      float z  = sigmoidf_(xz + aZ.f[v] + bhZ);
      float n  = tanhf_(xn + r * (aN.f[v] + bhN));
      float hn = (1.0f - z) * n + z * hReg[v];
      hReg[v]  = hn;
      unsigned short hb = f2bf(hn);
      hWr[(size_t)b * H_ + colh] = hb;
      seqOut[((size_t)b * T_ + t) * H_ + colh] = hb;
    }

    ++epoch;
    group_barrier(myctr, 16u * epoch);
  }

  #pragma unroll
  for (int v = 0; v < 8; ++v) {
    int b = g * 16 + v + 8 * u;
    hFinal[(size_t)b * H_ + colh] = hReg[v];
  }
}

// ----------------------------------------------------------- final FC (f32)
__global__ __launch_bounds__(256) void k_fc(const float* __restrict__ h,
                                            const float* __restrict__ w,
                                            const float* __restrict__ bias,
                                            float* __restrict__ out) {
  int b = blockIdx.x;
  const float4* hv = (const float4*)(h + (size_t)b * H_);
  for (int cc = threadIdx.x; cc < 1000; cc += blockDim.x) {
    const float4* wv = (const float4*)(w + (size_t)cc * H_);
    float acc = 0.f;
    #pragma unroll 4
    for (int k = 0; k < H_ / 4; ++k) {
      float4 a = hv[k], q = wv[k];
      acc = fmaf(a.x, q.x, fmaf(a.y, q.y, fmaf(a.z, q.z, fmaf(a.w, q.w, acc))));
    }
    out[(size_t)b * 1000 + cc] = acc + bias[cc];
  }
}

// ------------------------------------------------------------------- driver
extern "C" void kernel_launch(void* const* d_in, const int* in_sizes, int n_in,
                              void* d_out, int out_size, void* d_ws, size_t ws_size,
                              hipStream_t stream) {
  const float* x = (const float*)d_in[0];
  const float *Wih[4], *Whh[4], *bih[4], *bhh[4];
  for (int l = 0; l < 4; ++l) {
    Wih[l] = (const float*)d_in[1 + 4 * l + 0];
    Whh[l] = (const float*)d_in[1 + 4 * l + 1];
    bih[l] = (const float*)d_in[1 + 4 * l + 2];
    bhh[l] = (const float*)d_in[1 + 4 * l + 3];
  }
  const float* fcw = (const float*)d_in[17];
  const float* fcb = (const float*)d_in[18];
  float* out = (float*)d_out;

  // Workspace layout (~368.5 MiB total)
  char* ws = (char*)d_ws;
  size_t off = 0;
  auto alloc = [&](size_t bytes) -> void* {
    void* p = ws + off;
    off = (off + bytes + 255) & ~(size_t)255;
    return p;
  };
  unsigned*       ctr   = (unsigned*)alloc(16 * sizeof(unsigned));
  unsigned short* hping = (unsigned short*)alloc((size_t)2 * B_ * H_ * 2);
  float*          hfin  = (float*)alloc((size_t)B_ * H_ * 4);
  unsigned short* seqA  = (unsigned short*)alloc((size_t)MROWS * H_ * 2);
  unsigned short* seqB  = (unsigned short*)alloc((size_t)MROWS * H_ * 2);
  unsigned short* gx    = (unsigned short*)alloc((size_t)MROWS * G3H * 2);
  unsigned short* WihB  = (unsigned short*)alloc((size_t)4 * G3H * H_ * 2);
  unsigned short* WhhB  = (unsigned short*)alloc((size_t)4 * G3H * H_ * 2);

  hipLaunchKernelGGL(k_init_ctr, dim3(1), dim3(32), 0, stream, ctr);
  hipLaunchKernelGGL(k_f32_to_bf16, dim3(1024), dim3(256), 0, stream,
                     x, seqA, MROWS * I_);
  for (int l = 0; l < 4; ++l) {
    int K = (l == 0) ? I_ : H_;
    hipLaunchKernelGGL(k_f32_to_bf16, dim3(1024), dim3(256), 0, stream,
                       Wih[l], WihB + (size_t)l * G3H * H_, G3H * K);
    hipLaunchKernelGGL(k_f32_to_bf16, dim3(1024), dim3(256), 0, stream,
                       Whh[l], WhhB + (size_t)l * G3H * H_, G3H * H_);
  }

  unsigned short* bufs[2] = { seqA, seqB };
  for (int l = 0; l < 4; ++l) {
    int K = (l == 0) ? I_ : H_;
    unsigned short* inb  = bufs[l & 1];
    unsigned short* outb = bufs[(l + 1) & 1];
    hipLaunchKernelGGL(k_gemm_gx, dim3(MROWS / 128, G3H / 128), dim3(256), 0,
                       stream, inb, WihB + (size_t)l * G3H * H_, bih[l], gx, K);
    hipLaunchKernelGGL(k_gru_scan, dim3(64), dim3(128), 0, stream,
                       gx, WhhB + (size_t)l * G3H * H_, bhh[l],
                       hping, outb, hfin, ctr + l * 4);
  }
  hipLaunchKernelGGL(k_fc, dim3(B_), dim3(256), 0, stream, hfin, fcw, fcb, out);

  (void)in_sizes; (void)n_in; (void)out_size; (void)ws_size;
}
